// MaskDecoderLayer_3573412790602
// MI455X (gfx1250) — matrix operand order repaired and checked by reading, not compiled, not run on hardware
//
#include <hip/hip_runtime.h>
#include <hip/hip_bf16.h>

// ------------------------------------------------------------------
// MaskDecoderLayer for MI455X (gfx1250, wave32, WMMA bf16 path)
// Software-pipelined (double-buffered LDS) GEMM + flash attention.
// ------------------------------------------------------------------
constexpr int Bc = 4, Qc = 256, Lc = 4096, Dc = 256, Hc = 8, DHc = 32, GHc = 512;
constexpr float NEGF = -1e9f;
constexpr float MASK_THRESH = -2.1972245773362196f; // logit(0.1)

typedef __attribute__((ext_vector_type(16))) __bf16 v16bf;
typedef __attribute__((ext_vector_type(8)))  __bf16 v8bf;
typedef __attribute__((ext_vector_type(4)))  __bf16 v4bf;
typedef __attribute__((ext_vector_type(8)))  float  v8f;
typedef __attribute__((ext_vector_type(4)))  float  v4f;

__device__ __forceinline__ v8f wmma_bf16(v16bf a, v16bf b, v8f c) {
  // D = A(16x32 bf16) x B(32x16 bf16) + C(16x16 f32)
  return __builtin_amdgcn_wmma_f32_16x16x32_bf16(false, a, false, b, (short)0, c, false, false);
}

// A-fragment: element e -> k = ((e>>3)<<4) + 8*g + (e&7): two contiguous
// 8-element chunks at k=8g and k=16+8g of a row-major [m][k] LDS row.
__device__ __forceinline__ v16bf lds_a_frag(const __bf16* rowp, int g) {
  v8bf lo = *(const v8bf*)(rowp + 8 * g);
  v8bf hi = *(const v8bf*)(rowp + 16 + 8 * g);
  return __builtin_shufflevector(lo, hi, 0, 1, 2, 3, 4, 5, 6, 7, 8, 9, 10, 11, 12, 13, 14, 15);
}

// B-fragment: element e -> k = ((e>>3)<<4) + 8*g + (e&7) (n fixed): two 8-element
// chunks at k=8g and k=16+8g of a transposed [n][k] LDS row.
__device__ __forceinline__ v16bf lds_b_frag(const __bf16* rowp, int g) {
  v8bf lo = *(const v8bf*)(rowp + 8 * g);
  v8bf hi = *(const v8bf*)(rowp + 16 + 8 * g);
  return __builtin_shufflevector(lo, hi, 0, 1, 2, 3, 4, 5, 6, 7, 8, 9, 10, 11, 12, 13, 14, 15);
}

__device__ __forceinline__ v4bf cvt4(v4f f) {
  v4bf h;
  h[0] = (__bf16)f[0]; h[1] = (__bf16)f[1]; h[2] = (__bf16)f[2]; h[3] = (__bf16)f[3];
  return h;
}

// ------------------------------------------------------------------
// LayerNorm over D=256: one wave per row, 8 rows per block (float4 path)
// ------------------------------------------------------------------
__global__ __launch_bounds__(256) void ln_kernel(const float* __restrict__ x,
                                                 const float* __restrict__ gam,
                                                 const float* __restrict__ bet,
                                                 float* __restrict__ out, int rows) {
  int wave = threadIdx.x >> 5, lane = threadIdx.x & 31;
  int row = blockIdx.x * 8 + wave;
  if (row >= rows) return;
  const float* xr = x + (size_t)row * Dc + lane * 8;
  v4f a = *(const v4f*)xr;
  v4f b = *(const v4f*)(xr + 4);
  float s = 0.f, sq = 0.f;
#pragma unroll
  for (int i = 0; i < 4; i++) { s += a[i] + b[i]; sq += a[i] * a[i] + b[i] * b[i]; }
#pragma unroll
  for (int o = 16; o >= 1; o >>= 1) { s += __shfl_xor(s, o, 32); sq += __shfl_xor(sq, o, 32); }
  float mean = s * (1.f / Dc);
  float var  = sq * (1.f / Dc) - mean * mean;
  float inv  = rsqrtf(var + 1e-5f);
  v4f g0 = *(const v4f*)(gam + lane * 8), g1 = *(const v4f*)(gam + lane * 8 + 4);
  v4f b0 = *(const v4f*)(bet + lane * 8), b1 = *(const v4f*)(bet + lane * 8 + 4);
  v4f r0, r1;
#pragma unroll
  for (int i = 0; i < 4; i++) {
    r0[i] = (a[i] - mean) * inv * g0[i] + b0[i];
    r1[i] = (b[i] - mean) * inv * g1[i] + b1[i];
  }
  float* orow = out + (size_t)row * Dc + lane * 8;
  *(v4f*)orow = r0;
  *(v4f*)(orow + 4) = r1;
}

// ------------------------------------------------------------------
// Generic WMMA GEMM: C = A[M,K] @ B[K,N] (+bias) (+residual)
// Block tile 128x64, 8 waves (4x2), wave tile 32x32, K-step 32.
// Double-buffered LDS; global loads of tile i+1 issued before the WMMAs
// of tile i, converted/committed after them. One barrier per K-step.
// ------------------------------------------------------------------
__global__ __launch_bounds__(256) void gemm_kernel(
    const float* __restrict__ A, long long sAb,
    const float* __restrict__ Bw, long long sBb, int transB,
    const float* __restrict__ bias,
    const float* R, long long sRb,
    float* C, long long sCb,
    int M, int N, int K)
{
  __shared__ alignas(16) __bf16 As[2][128][40];
  __shared__ alignas(16) __bf16 Bt[2][64][40];
  int bz = blockIdx.z;
  A  += (size_t)bz * sAb;
  Bw += (size_t)bz * sBb;
  C  += (size_t)bz * sCb;
  if (R) R += (size_t)bz * sRb;

  int m0 = blockIdx.y * 128, n0 = blockIdx.x * 64;
  int tid = threadIdx.x, lane = tid & 31, wave = tid >> 5;
  int wm = wave >> 1, wn = wave & 1;
  int g = lane >> 4, l15 = lane & 15;

  v4f ra[4], rb[2];
  auto issue = [&](int k0) {
#pragma unroll
    for (int i = 0; i < 4; i++) {
      int idx = tid + i * 256; int r = idx >> 3, c4 = (idx & 7) << 2;
      ra[i] = *(const v4f*)&A[(size_t)(m0 + r) * K + k0 + c4];
    }
    if (!transB) {
#pragma unroll
      for (int i = 0; i < 2; i++) {
        int idx = tid + i * 256; int r = idx >> 4, c4 = (idx & 15) << 2;
        rb[i] = *(const v4f*)&Bw[(size_t)(k0 + r) * N + n0 + c4];
      }
    } else {
#pragma unroll
      for (int i = 0; i < 2; i++) {
        int idx = tid + i * 256; int n = idx >> 3, k4 = (idx & 7) << 2;
        rb[i] = *(const v4f*)&Bw[(size_t)(n0 + n) * K + k0 + k4];
      }
    }
  };
  auto commit = [&](int buf) {
#pragma unroll
    for (int i = 0; i < 4; i++) {
      int idx = tid + i * 256; int r = idx >> 3, c4 = (idx & 7) << 2;
      *(v4bf*)&As[buf][r][c4] = cvt4(ra[i]);
    }
    if (!transB) {
#pragma unroll
      for (int i = 0; i < 2; i++) {
        int idx = tid + i * 256; int r = idx >> 4, c4 = (idx & 15) << 2;
        Bt[buf][c4 + 0][r] = (__bf16)rb[i][0];
        Bt[buf][c4 + 1][r] = (__bf16)rb[i][1];
        Bt[buf][c4 + 2][r] = (__bf16)rb[i][2];
        Bt[buf][c4 + 3][r] = (__bf16)rb[i][3];
      }
    } else {
#pragma unroll
      for (int i = 0; i < 2; i++) {
        int idx = tid + i * 256; int n = idx >> 3, k4 = (idx & 7) << 2;
        *(v4bf*)&Bt[buf][n][k4] = cvt4(rb[i]);
      }
    }
  };

  v8f acc[2][2];
#pragma unroll
  for (int i = 0; i < 2; i++)
#pragma unroll
    for (int j = 0; j < 2; j++)
#pragma unroll
      for (int r = 0; r < 8; r++) acc[i][j][r] = 0.f;

  issue(0);
  commit(0);
  __syncthreads();
  int cur = 0;
  for (int k0 = 0; k0 < K; k0 += 32) {
    bool nxt = (k0 + 32) < K;
    if (nxt) issue(k0 + 32);                 // loads in flight during WMMAs
    if (k0 + 64 < K)                          // warm L2 two tiles ahead
      __builtin_prefetch(&A[(size_t)(m0 + (tid >> 3)) * K + k0 + 64 + ((tid & 7) << 2)], 0, 1);

    v16bf af[2], bfg[2];
#pragma unroll
    for (int sm = 0; sm < 2; sm++) af[sm]  = lds_a_frag(&As[cur][wm * 32 + sm * 16 + l15][0], g);
#pragma unroll
    for (int sn = 0; sn < 2; sn++) bfg[sn] = lds_b_frag(&Bt[cur][wn * 32 + sn * 16 + l15][0], g);
#pragma unroll
    for (int sm = 0; sm < 2; sm++)
#pragma unroll
      for (int sn = 0; sn < 2; sn++)
        acc[sm][sn] = wmma_bf16(af[sm], bfg[sn], acc[sm][sn]);

    if (nxt) commit(cur ^ 1);                // waits loads after compute
    __syncthreads();
    cur ^= 1;
  }

  // epilogue: bias + residual, C row layout r + 8*g
#pragma unroll
  for (int sm = 0; sm < 2; sm++)
#pragma unroll
    for (int sn = 0; sn < 2; sn++)
#pragma unroll
      for (int r = 0; r < 8; r++) {
        int m = m0 + wm * 32 + sm * 16 + r + 8 * g;
        int n = n0 + wn * 32 + sn * 16 + l15;
        float v = acc[sm][sn][r];
        if (bias) v += bias[n];
        if (R) v += R[(size_t)m * N + n];
        C[(size_t)m * N + n] = v;
      }
}

// ------------------------------------------------------------------
// Mask build: blocked = sigmoid(score)<0.1; fully-blocked rows unblock
// one block per (b,q) row; thread t owns 16 consecutive elements
// ------------------------------------------------------------------
__global__ __launch_bounds__(256) void mask_kernel(const float* __restrict__ scores,
                                                   unsigned char* __restrict__ am) {
  __shared__ int allb;
  if (threadIdx.x == 0) allb = 1;
  __syncthreads();
  const float* row = scores + (size_t)blockIdx.x * Lc + threadIdx.x * 16;
  unsigned int flags[4]; int myall = 1;
#pragma unroll
  for (int i = 0; i < 4; i++) {
    v4f f = *(const v4f*)(row + i * 4);
    unsigned int w = 0;
#pragma unroll
    for (int j = 0; j < 4; j++) {
      unsigned int bl = (f[j] < MASK_THRESH) ? 1u : 0u;
      myall &= (int)bl;
      w |= bl << (8 * j);
    }
    flags[i] = w;
  }
  if (!myall) atomicAnd(&allb, 0);
  __syncthreads();
  unsigned int fix = allb ? 0u : 0x01010101u; // all blocked -> unblock everything
  unsigned int* arow = (unsigned int*)(am + (size_t)blockIdx.x * Lc + threadIdx.x * 16);
#pragma unroll
  for (int i = 0; i < 4; i++) arow[i] = flags[i] & fix;
}

// ------------------------------------------------------------------
// Flash attention (DH=32). 4 waves x 16 q-rows, k-tiles of 64.
// Ks: [kvrow][dh] (QK^T B-frags contiguous); Vt: [dh][kvrow] transposed
// (PV B-frags contiguous); Ps: [wave][qrow][k] (PV A-frags contiguous).
// Double-buffered K/V tiles; one barrier per k-tile. Ps is wave-private
// (LDS is in-order per wave, no barrier needed between store and reload).
// mask[b*msB + row*msR + col*msC] != 0  => blocked (NEG)
// ------------------------------------------------------------------
__global__ __launch_bounds__(128) void attn_kernel(
    const float* __restrict__ Qm, const float* __restrict__ Km, const float* __restrict__ Vm,
    float* __restrict__ O,
    const unsigned char* __restrict__ mask, long long msB, long long msR, long long msC,
    const unsigned char* __restrict__ kvmask,
    int Sq, int Sk)
{
  __shared__ alignas(16) __bf16 Ks[2][64][40];
  __shared__ alignas(16) __bf16 Vt[2][32][72];
  __shared__ alignas(16) __bf16 Ps[4][16][72];

  int b = blockIdx.z, h = blockIdx.y;
  int tid = threadIdx.x, lane = tid & 31, wave = tid >> 5;
  int g = lane >> 4, l15 = lane & 15;
  int q0 = blockIdx.x * 64 + wave * 16;
  const float scale = 0.17677669529663688f; // 1/sqrt(32)

  v4f rk[4], rv[4];
  auto issue = [&](int kt) {
#pragma unroll
    for (int i = 0; i < 4; i++) {
      int idx = tid + i * 128; int r = idx >> 3, c4 = (idx & 7) << 2;
      size_t off = ((size_t)b * Sk + kt + r) * Dc + h * DHc + c4;
      rk[i] = *(const v4f*)(Km + off);
      rv[i] = *(const v4f*)(Vm + off);
    }
  };
  auto commit = [&](int buf) {
#pragma unroll
    for (int i = 0; i < 4; i++) {
      int idx = tid + i * 128; int r = idx >> 3, c4 = (idx & 7) << 2;
      *(v4bf*)&Ks[buf][r][c4] = cvt4(rk[i]);
      Vt[buf][c4 + 0][r] = (__bf16)rv[i][0];
      Vt[buf][c4 + 1][r] = (__bf16)rv[i][1];
      Vt[buf][c4 + 2][r] = (__bf16)rv[i][2];
      Vt[buf][c4 + 3][r] = (__bf16)rv[i][3];
    }
  };

  // Q fragment (A layout) loaded once, via 16B global loads
  v16bf aq;
  {
    const float* qr = Qm + ((size_t)b * Sq + q0 + l15) * Dc + h * DHc;
    v4f f0 = *(const v4f*)(qr + 8 * g);
    v4f f1 = *(const v4f*)(qr + 8 * g + 4);
    v4f f2 = *(const v4f*)(qr + 16 + 8 * g);
    v4f f3 = *(const v4f*)(qr + 16 + 8 * g + 4);
#pragma unroll
    for (int i = 0; i < 4; i++) {
      aq[i]      = (__bf16)f0[i];
      aq[4 + i]  = (__bf16)f1[i];
      aq[8 + i]  = (__bf16)f2[i];
      aq[12 + i] = (__bf16)f3[i];
    }
  }

  float mrow[8], lsum[8];
#pragma unroll
  for (int r = 0; r < 8; r++) { mrow[r] = -3.0e38f; lsum[r] = 0.f; }
  v8f o0, o1;
#pragma unroll
  for (int r = 0; r < 8; r++) { o0[r] = 0.f; o1[r] = 0.f; }

  issue(0);
  commit(0);
  __syncthreads();
  int cur = 0;
  for (int kt = 0; kt < Sk; kt += 64) {
    bool nxt = (kt + 64) < Sk;
    if (nxt) issue(kt + 64);                  // next K/V loads overlap compute

    // S = Q @ K^T : four 16x16 tiles (K=32, one WMMA each)
    v8f s[4];
#pragma unroll
    for (int j = 0; j < 4; j++) {
      v16bf bk = lds_b_frag(&Ks[cur][j * 16 + l15][0], g);
      v8f z;
#pragma unroll
      for (int r = 0; r < 8; r++) z[r] = 0.f;
      s[j] = wmma_bf16(aq, bk, z);
    }

    // scale + masks
#pragma unroll
    for (int j = 0; j < 4; j++) {
      int col = kt + j * 16 + l15;
      bool kvok = (kvmask == nullptr) || (kvmask[(size_t)b * Sk + col] != 0);
#pragma unroll
      for (int r = 0; r < 8; r++) {
        float v = s[j][r] * scale;
        int row = q0 + r + 8 * g;
        if (mask && mask[(size_t)b * msB + (size_t)row * msR + (size_t)col * msC]) v = NEGF;
        if (!kvok) v = NEGF;
        s[j][r] = v;
      }
    }

    // online softmax per row (rows live in 16-lane halves)
    float alpha[8];
#pragma unroll
    for (int r = 0; r < 8; r++) {
      float mx = s[0][r];
#pragma unroll
      for (int j = 1; j < 4; j++) mx = fmaxf(mx, s[j][r]);
#pragma unroll
      for (int o = 8; o >= 1; o >>= 1) mx = fmaxf(mx, __shfl_xor(mx, o, 32));
      float nm = fmaxf(mrow[r], mx);
      alpha[r] = expf(mrow[r] - nm);
      mrow[r] = nm;
      float ps = 0.f;
#pragma unroll
      for (int j = 0; j < 4; j++) { float p = expf(s[j][r] - nm); s[j][r] = p; ps += p; }
#pragma unroll
      for (int o = 8; o >= 1; o >>= 1) ps += __shfl_xor(ps, o, 32);
      lsum[r] = lsum[r] * alpha[r] + ps;
      o0[r] *= alpha[r];
      o1[r] *= alpha[r];
    }

    // P -> LDS (C-layout scatter), wave-private region
#pragma unroll
    for (int j = 0; j < 4; j++)
#pragma unroll
      for (int r = 0; r < 8; r++)
        Ps[wave][r + 8 * g][j * 16 + l15] = (__bf16)s[j][r];

    // O += P(16x64) @ V(64x32): 2 K-steps x 2 N-subtiles
#pragma unroll
    for (int ks = 0; ks < 2; ks++) {
      v16bf ap  = lds_a_frag(&Ps[wave][l15][ks * 32], g);
      v16bf bv0 = lds_b_frag(&Vt[cur][l15][ks * 32], g);
      v16bf bv1 = lds_b_frag(&Vt[cur][16 + l15][ks * 32], g);
      o0 = wmma_bf16(ap, bv0, o0);
      o1 = wmma_bf16(ap, bv1, o1);
    }

    if (nxt) commit(cur ^ 1);                 // waits loads after compute
    __syncthreads();
    cur ^= 1;
  }

#pragma unroll
  for (int r = 0; r < 8; r++) {
    float inv = 1.0f / lsum[r];
    int row = q0 + r + 8 * g;
    float* orow = O + ((size_t)b * Sq + row) * Dc + h * DHc;
    orow[l15]      = o0[r] * inv;
    orow[16 + l15] = o1[r] * inv;
  }
}

// ------------------------------------------------------------------
// GLU activation: g = u[:, :GH] * silu(u[:, GH:]) (float4 path)
// ------------------------------------------------------------------
__global__ __launch_bounds__(256) void glu_act_kernel(const float* __restrict__ u,
                                                      float* __restrict__ gout, int rows) {
  size_t i = (size_t)blockIdx.x * 256 + threadIdx.x;   // i indexes float4 chunks
  size_t n4 = (size_t)rows * (GHc / 4);
  if (i >= n4) return;
  size_t row = i / (GHc / 4), c4 = (i % (GHc / 4)) * 4;
  v4f a  = *(const v4f*)&u[row * (2 * GHc) + c4];
  v4f bb = *(const v4f*)&u[row * (2 * GHc) + GHc + c4];
  v4f r;
#pragma unroll
  for (int j = 0; j < 4; j++) r[j] = a[j] * (bb[j] / (1.f + expf(-bb[j])));
  *(v4f*)&gout[row * GHc + c4] = r;
}

// ------------------------------------------------------------------
// Workspace layout (bytes) -- carefully overlapped across phases
// ------------------------------------------------------------------
constexpr size_t SZ_BQL_F = (size_t)Bc * Qc * Lc * 4;  // 16 MB
constexpr size_t SZ_BQL_U = (size_t)Bc * Qc * Lc;      //  4 MB
constexpr size_t SZ_BLD   = (size_t)Bc * Lc * Dc * 4;  // 16 MB
constexpr size_t SZ_BQD   = (size_t)Bc * Qc * Dc * 4;  //  1 MB
constexpr size_t OFF_SCORES = 0;
constexpr size_t OFF_AM  = OFF_SCORES + SZ_BQL_F;
constexpr size_t OFF_LNA = OFF_AM + SZ_BQL_U;
constexpr size_t OFF_LNB = OFF_LNA + SZ_BLD;
constexpr size_t OFF_X1  = OFF_LNB + SZ_BQD;
constexpr size_t OFF_X2  = OFF_X1 + SZ_BLD;
constexpr size_t OFF_X3  = OFF_X2 + SZ_BLD;
constexpr size_t OFF_AO  = OFF_X3 + SZ_BLD;
constexpr size_t OFF_MT  = OFF_AO + SZ_BLD;
constexpr size_t OFF_U   = OFF_X1;   // 64 MB, spans X1..AO (dead during GLU)
constexpr size_t OFF_G   = 0;        // 32 MB, spans SCORES/AM/LNA(part) (dead then)

static void launch_gemm(hipStream_t s, const float* A, long long sA,
                        const float* Bw, long long sB, int transB,
                        const float* bias, const float* R, long long sR,
                        float* C, long long sC, int M, int N, int K, int batch) {
  dim3 grid(N / 64, M / 128, batch);
  gemm_kernel<<<grid, dim3(256), 0, s>>>(A, sA, Bw, sB, transB, bias, R, sR, C, sC, M, N, K);
}

extern "C" void kernel_launch(void* const* d_in, const int* in_sizes, int n_in,
                              void* d_out, int out_size, void* d_ws, size_t ws_size,
                              hipStream_t stream) {
  (void)in_sizes; (void)n_in; (void)out_size; (void)ws_size;
  const float* q_in  = (const float*)d_in[0];
  const float* kv_in = (const float*)d_in[1];
  const unsigned char* kvmask = (const unsigned char*)d_in[2];
  auto P = [&](int i) { return (const float*)d_in[i]; };
  // params: 3 mask_w, 4 mask_b,
  // q_ca:  5 wq 6 bq 7 wk 8 bk 9 wv 10 bv 11 wo 12 bo 13 lnq_g 14 lnq_b 15 lnkv_g 16 lnkv_b
  // q_sa: 17 wq 18 bq 19 wk 20 bk 21 wv 22 bv 23 wo 24 bo 25 ln_g 26 ln_b
  // q_glu: 27 ln_g 28 ln_b 29 w1 30 b1 31 w2 32 b2
  // kv_ca: 33..40 attn, 41 lnq_g 42 lnq_b 43 lnkv_g 44 lnkv_b
  // kv_glu: 45 ln_g 46 ln_b 47 w1 48 b1 49 w2 50 b2

  char* ws = (char*)d_ws;
  float* scores = (float*)(ws + OFF_SCORES);
  unsigned char* am = (unsigned char*)(ws + OFF_AM);
  float* lnA = (float*)(ws + OFF_LNA);
  float* lnB = (float*)(ws + OFF_LNB);
  float* X1  = (float*)(ws + OFF_X1);
  float* X2  = (float*)(ws + OFF_X2);
  float* X3  = (float*)(ws + OFF_X3);
  float* AO  = (float*)(ws + OFF_AO);
  float* MT  = (float*)(ws + OFF_MT);
  float* U   = (float*)(ws + OFF_U);
  float* G   = (float*)(ws + OFF_G);

  const int BQ = Bc * Qc, BL = Bc * Lc;
  float* q_out  = (float*)d_out;
  float* kv_out = q_out + (size_t)BQ * Dc;

  // residual streams live in d_out
  hipMemcpyAsync(q_out,  q_in,  (size_t)BQ * Dc * 4, hipMemcpyDeviceToDevice, stream);
  hipMemcpyAsync(kv_out, kv_in, (size_t)BL * Dc * 4, hipMemcpyDeviceToDevice, stream);

  // 1) mask tokens: mt = q @ mask_w + mask_b
  launch_gemm(stream, q_in, 0, P(3), 0, 0, P(4), nullptr, 0, MT, 0, BQ, Dc, Dc, 1);
  // 2) scores[b] = mt[b] @ kv[b]^T   (transB, batched)
  launch_gemm(stream, MT, (long long)Qc * Dc, kv_in, (long long)Lc * Dc, 1,
              nullptr, nullptr, 0, scores, (long long)Qc * Lc, Qc, Lc, Dc, Bc);
  // 3) am = blocked mask with fully-blocked-row fix
  mask_kernel<<<dim3(Bc * Qc), dim3(256), 0, stream>>>(scores, am);

  // 4) q cross-attention (q attends to kv)
  ln_kernel<<<dim3(BQ / 8), dim3(256), 0, stream>>>(q_in,  P(13), P(14), lnB, BQ);
  ln_kernel<<<dim3(BL / 8), dim3(256), 0, stream>>>(kv_in, P(15), P(16), lnA, BL);
  launch_gemm(stream, lnB, 0, P(5), 0, 0, P(6),  nullptr, 0, X1, 0, BQ, Dc, Dc, 1);
  launch_gemm(stream, lnA, 0, P(7), 0, 0, P(8),  nullptr, 0, X2, 0, BL, Dc, Dc, 1);
  launch_gemm(stream, lnA, 0, P(9), 0, 0, P(10), nullptr, 0, X3, 0, BL, Dc, Dc, 1);
  attn_kernel<<<dim3(Qc / 64, Hc, Bc), dim3(128), 0, stream>>>(
      X1, X2, X3, AO, am, (long long)Qc * Lc, (long long)Lc, 1LL, kvmask, Qc, Lc);
  launch_gemm(stream, AO, 0, P(11), 0, 0, P(12), q_out, 0, q_out, 0, BQ, Dc, Dc, 1);

  // 5) q self-attention
  ln_kernel<<<dim3(BQ / 8), dim3(256), 0, stream>>>(q_out, P(25), P(26), lnB, BQ);
  launch_gemm(stream, lnB, 0, P(17), 0, 0, P(18), nullptr, 0, X1, 0, BQ, Dc, Dc, 1);
  launch_gemm(stream, lnB, 0, P(19), 0, 0, P(20), nullptr, 0, X2, 0, BQ, Dc, Dc, 1);
  launch_gemm(stream, lnB, 0, P(21), 0, 0, P(22), nullptr, 0, X3, 0, BQ, Dc, Dc, 1);
  attn_kernel<<<dim3(Qc / 64, Hc, Bc), dim3(128), 0, stream>>>(
      X1, X2, X3, AO, nullptr, 0, 0, 0, nullptr, Qc, Qc);
  launch_gemm(stream, AO, 0, P(23), 0, 0, P(24), q_out, 0, q_out, 0, BQ, Dc, Dc, 1);

  // 6) q GLU
  ln_kernel<<<dim3(BQ / 8), dim3(256), 0, stream>>>(q_out, P(27), P(28), lnB, BQ);
  launch_gemm(stream, lnB, 0, P(29), 0, 0, P(30), nullptr, 0, U, 0, BQ, 2 * GHc, Dc, 1);
  glu_act_kernel<<<dim3((BQ * (GHc / 4) + 255) / 256), dim3(256), 0, stream>>>(U, G, BQ);
  launch_gemm(stream, G, 0, P(31), 0, 0, P(32), q_out, 0, q_out, 0, BQ, Dc, GHc, 1);

  // 7) kv cross-attention (kv attends to updated q), mask transposed
  ln_kernel<<<dim3(BL / 8), dim3(256), 0, stream>>>(kv_in, P(41), P(42), lnA, BL);
  ln_kernel<<<dim3(BQ / 8), dim3(256), 0, stream>>>(q_out, P(43), P(44), lnB, BQ);
  launch_gemm(stream, lnA, 0, P(33), 0, 0, P(34), nullptr, 0, X1, 0, BL, Dc, Dc, 1);
  launch_gemm(stream, lnB, 0, P(35), 0, 0, P(36), nullptr, 0, X2, 0, BQ, Dc, Dc, 1);
  launch_gemm(stream, lnB, 0, P(37), 0, 0, P(38), nullptr, 0, X3, 0, BQ, Dc, Dc, 1);
  attn_kernel<<<dim3(Lc / 64, Hc, Bc), dim3(128), 0, stream>>>(
      X1, X2, X3, AO, am, (long long)Qc * Lc, 1LL, (long long)Lc, nullptr, Lc, Qc);
  launch_gemm(stream, AO, 0, P(39), 0, 0, P(40), kv_out, 0, kv_out, 0, BL, Dc, Dc, 1);

  // 8) kv GLU
  ln_kernel<<<dim3(BL / 8), dim3(256), 0, stream>>>(kv_out, P(45), P(46), lnA, BL);
  launch_gemm(stream, lnA, 0, P(47), 0, 0, P(48), nullptr, 0, U, 0, BL, 2 * GHc, Dc, 1);
  glu_act_kernel<<<dim3((BL * (GHc / 4) + 255) / 256), dim3(256), 0, stream>>>(U, G, BL);
  launch_gemm(stream, G, 0, P(49), 0, 0, P(50), kv_out, 0, kv_out, 0, BL, Dc, GHc, 1);
}
